// OptimizedPagedMultiLatentAttention_45354854645801
// MI455X (gfx1250) — matrix operand order repaired and checked by reading, not compiled, not run on hardware
//
#include <hip/hip_runtime.h>
#include <hip/hip_bf16.h>

// ---------------------------------------------------------------------------
// CDNA5 (gfx1250) WMMA types
// ---------------------------------------------------------------------------
typedef __attribute__((ext_vector_type(16))) __bf16 v16bf;
typedef __attribute__((ext_vector_type(8)))  __bf16 v8bf;
typedef __attribute__((ext_vector_type(8)))  float  v8f;

#define WMMA_BF16(a, b, c) \
    __builtin_amdgcn_wmma_f32_16x16x32_bf16(false, (a), false, (b), (short)0, (c), false, false)

// Problem constants (from reference): B=2, T=2048, E=1024, H=16, D=64, L=512
#define C_B    2
#define C_T    2048
#define C_E    1024
#define C_H    16
#define C_D    64
#define C_L    512
#define C_MTOT (C_B * C_T)   // 4096
#define C_PAGE 256

// ---------------------------------------------------------------------------
// Fragment builders (16-bit; contiguous k order used by this code, not the ISA table order)
//
// A (16x32, M x K): lane = hf*16 + m (m = row).  Element i of v16bf:
//   K = hf*16 + i                  (contiguous run of 16)
//
// B (32x16, K x N): lane = hf*16 + n (n = col).  Element i of v16bf:
//   K = hf*16 + i                  (contiguous run of 16)
// ---------------------------------------------------------------------------
static __device__ __forceinline__ v16bf make_afrag(const __bf16* row, int hf) {
    v8bf r1 = *(const v8bf*)(row + hf * 16);
    v8bf r2 = *(const v8bf*)(row + hf * 16 + 8);
    v16bf a;
#pragma unroll
    for (int i = 0; i < 8; ++i) { a[i] = r1[i]; a[i + 8] = r2[i]; }
    return a;
}

static __device__ __forceinline__ v16bf ld_contig16(const __bf16* p) {
    v8bf r1 = *(const v8bf*)(p);
    v8bf r2 = *(const v8bf*)(p + 8);
    v16bf b;
#pragma unroll
    for (int i = 0; i < 8; ++i) { b[i] = r1[i]; b[i + 8] = r2[i]; }
    return b;
}

// ---------------------------------------------------------------------------
// fp32 -> bf16 conversion (grid-stride)
// ---------------------------------------------------------------------------
__global__ void cvt_f32_to_bf16(const float* __restrict__ src, __bf16* __restrict__ dst, int n) {
    int i = blockIdx.x * blockDim.x + threadIdx.x;
    int stride = gridDim.x * blockDim.x;
    for (; i < n; i += stride) dst[i] = (__bf16)src[i];
}

// ---------------------------------------------------------------------------
// RoPE in-place on bf16 [Mtot, E] viewed as [Mtot, H, D]; t = row % T.
// Pairs are (2i, 2i+1) within each head, angle = t * theta^(-2i/D).
// ---------------------------------------------------------------------------
__global__ void rope_inplace(__bf16* __restrict__ buf) {
    const int halfE = C_E / 2;
    int idx = blockIdx.x * blockDim.x + threadIdx.x;
    if (idx >= C_MTOT * halfE) return;
    int row = idx / halfE;
    int pc  = idx - row * halfE;
    int h   = pc / (C_D / 2);
    int i   = pc - h * (C_D / 2);
    int t   = row % C_T;
    // theta^(-2i/D) = exp(-(2i/D) * ln(10000))
    float inv = __expf(-(2.0f * (float)i / (float)C_D) * 9.210340371976184f);
    float ang = (float)t * inv;
    float s, c;
    __sincosf(ang, &s, &c);
    __bf16* p = buf + (size_t)row * C_E + h * C_D + 2 * i;
    float x1 = (float)p[0];
    float x2 = (float)p[1];
    p[0] = (__bf16)(x1 * c - x2 * s);
    p[1] = (__bf16)(x1 * s + x2 * c);
}

// ---------------------------------------------------------------------------
// Tiled WMMA GEMM: C[M,N] = A[M,K] @ B[K,N], bf16 inputs, fp32 accumulate.
// Block = 128 threads (4 waves). Block tile 128x64, K-step 32.
// Each wave owns a 32x64 sub-tile (2 M-tiles x 4 N-tiles = 8 WMMAs/K-step),
// reusing the 4 B-fragments across both M-tiles.  All fragments are
// preloaded before the WMMA group so the 8 WMMAs issue back-to-back after a
// single LDS wait.
// mode 0: store bf16 row-major C[M,N]
// mode 1: store bf16 TRANSPOSED C^T[N,M]  (packed v8bf stores)
// mode 2: store f32  row-major C[M,N]
// ---------------------------------------------------------------------------
__global__ void __launch_bounds__(128)
gemm_bf16_wmma(const __bf16* __restrict__ A, const __bf16* __restrict__ B,
               void* __restrict__ C, int M, int N, int K, int mode) {
    __shared__ __bf16 sA[128][32];   // A tile, row-major [m][k]   (8 KB)
    __shared__ __bf16 sBT[64][32];   // B tile transposed [n][k]   (4 KB)

    const int tid  = threadIdx.x;
    const int wv   = tid >> 5;
    const int lane = tid & 31;
    const int hf   = lane >> 4;
    const int l16  = lane & 15;
    const int m0   = blockIdx.y * 128;
    const int n0   = blockIdx.x * 64;

    const int arow = tid >> 1;          // 0..63 (and +64)
    const int acol = (tid & 1) * 16;    // 0 or 16
    const int brow = tid >> 2;          // 0..31
    const int bcol = (tid & 3) * 16;    // 0,16,32,48

    v8f acc[2][4] = {};

    for (int k0 = 0; k0 < K; k0 += 32) {
        // --- cooperative A tile load (two 64-row halves, coalesced b128) ---
        const __bf16* ap0 = A + (size_t)(m0 + arow) * K + k0 + acol;
        const __bf16* ap1 = A + (size_t)(m0 + 64 + arow) * K + k0 + acol;
        v8bf a0 = *(const v8bf*)(ap0);
        v8bf a1 = *(const v8bf*)(ap0 + 8);
        v8bf a2 = *(const v8bf*)(ap1);
        v8bf a3 = *(const v8bf*)(ap1 + 8);
        *(v8bf*)&sA[arow][acol]          = a0;
        *(v8bf*)&sA[arow][acol + 8]      = a1;
        *(v8bf*)&sA[64 + arow][acol]     = a2;
        *(v8bf*)&sA[64 + arow][acol + 8] = a3;

        // --- cooperative B tile load, transposed into LDS ---
        const __bf16* bp = B + (size_t)(k0 + brow) * N + n0 + bcol;
        v8bf b0 = *(const v8bf*)(bp);
        v8bf b1 = *(const v8bf*)(bp + 8);
#pragma unroll
        for (int j = 0; j < 8; ++j) {
            sBT[bcol + j][brow]     = b0[j];
            sBT[bcol + 8 + j][brow] = b1[j];
        }

        // prefetch next K tiles into cache while we compute
        if (k0 + 32 < K) {
            __builtin_prefetch(A + (size_t)(m0 + arow) * K + k0 + 32 + acol, 0, 3);
            __builtin_prefetch(A + (size_t)(m0 + 64 + arow) * K + k0 + 32 + acol, 0, 3);
            __builtin_prefetch(B + (size_t)(k0 + 32 + brow) * N + n0 + bcol, 0, 3);
        }

        __syncthreads();

        // --- preload ALL fragments, then fire 8 WMMAs back-to-back ---
        v16bf af0 = make_afrag(&sA[wv * 32 + l16][0], hf);
        v16bf af1 = make_afrag(&sA[wv * 32 + 16 + l16][0], hf);
        v16bf bfr[4];
#pragma unroll
        for (int nt = 0; nt < 4; ++nt)
            bfr[nt] = ld_contig16(&sBT[nt * 16 + l16][hf * 16]);

#pragma unroll
        for (int nt = 0; nt < 4; ++nt)
            acc[0][nt] = WMMA_BF16(af0, bfr[nt], acc[0][nt]);
#pragma unroll
        for (int nt = 0; nt < 4; ++nt)
            acc[1][nt] = WMMA_BF16(af1, bfr[nt], acc[1][nt]);

        __syncthreads();
    }

    // C/D layout: lane holds col n = l16, rows m = hf*8 + v (v = 0..7)
#pragma unroll
    for (int mt = 0; mt < 2; ++mt) {
        const int mrow = m0 + wv * 32 + mt * 16 + hf * 8;
        if (mode == 0) {
            __bf16* Cb = (__bf16*)C;
#pragma unroll
            for (int nt = 0; nt < 4; ++nt)
#pragma unroll
                for (int v = 0; v < 8; ++v)
                    Cb[(size_t)(mrow + v) * N + n0 + nt * 16 + l16] = (__bf16)acc[mt][nt][v];
        } else if (mode == 1) {
            __bf16* Cb = (__bf16*)C;   // C^T, shape [N][M]
#pragma unroll
            for (int nt = 0; nt < 4; ++nt) {
                v8bf pk;
#pragma unroll
                for (int v = 0; v < 8; ++v) pk[v] = (__bf16)acc[mt][nt][v];
                *(v8bf*)(Cb + (size_t)(n0 + nt * 16 + l16) * M + mrow) = pk;
            }
        } else {
            float* Cf = (float*)C;
#pragma unroll
            for (int nt = 0; nt < 4; ++nt)
#pragma unroll
                for (int v = 0; v < 8; ++v)
                    Cf[(size_t)(mrow + v) * N + n0 + nt * 16 + l16] = acc[mt][nt][v];
        }
    }
}

// ---------------------------------------------------------------------------
// Flash attention, one wave per (16-query tile, head, batch).
// Page-causal mask is block-granular (PAGE=256), so key range is simply
// [0, (qpage+1)*256) with no element masking.  Q/K row-major bf16 [Mtot,E];
// V pre-transposed bf16 [E, Mtot] so the P@V B-fragment is a contiguous load.
// All fragments for each WMMA group are preloaded so the WMMAs issue
// back-to-back after one wait.
// ---------------------------------------------------------------------------
__global__ void __launch_bounds__(32)
flash_attn_wmma(const __bf16* __restrict__ Q, const __bf16* __restrict__ Kb,
                const __bf16* __restrict__ VT, __bf16* __restrict__ O) {
    const int qt   = blockIdx.x;
    const int h    = blockIdx.y;
    const int b    = blockIdx.z;
    const int lane = threadIdx.x & 31;
    const int hf   = lane >> 4;
    const int l16  = lane & 15;
    const int q0   = qt * 16;
    const int rb   = b * C_T;

    // Q A-fragments: row = query token, contraction = d (two K=32 chunks)
    const __bf16* qrow = Q + (size_t)(rb + q0 + l16) * C_E + h * C_D;
    const v16bf a_lo = make_afrag(qrow, hf);        // d 0..31
    const v16bf a_hi = make_afrag(qrow + 32, hf);   // d 32..63

    float m_run[8], l_run[8];
    v8f acc[4] = {};
#pragma unroll
    for (int v = 0; v < 8; ++v) { m_run[v] = -3.0e38f; l_run[v] = 0.0f; }

    __shared__ __bf16 sP[16][32];   // P tile bounce: [query][key-in-block]

    const int kend = ((q0 >> 8) + 1) << 8;   // (qpage+1)*256
    for (int kv0 = 0; kv0 < kend; kv0 += 32) {
        // ----- S = Q @ K^T : preload 4 K-fragments, then 4 WMMAs
        const __bf16* kr0 = Kb + (size_t)(rb + kv0 + l16) * C_E + h * C_D;
        const __bf16* kr1 = Kb + (size_t)(rb + kv0 + 16 + l16) * C_E + h * C_D;
        v16bf kf0 = ld_contig16(kr0 + hf * 16);
        v16bf kf1 = ld_contig16(kr0 + 32 + hf * 16);
        v16bf kf2 = ld_contig16(kr1 + hf * 16);
        v16bf kf3 = ld_contig16(kr1 + 32 + hf * 16);
        v8f s0 = {}, s1 = {};
        s0 = WMMA_BF16(a_lo, kf0, s0);
        s0 = WMMA_BF16(a_hi, kf1, s0);
        s1 = WMMA_BF16(a_lo, kf2, s1);
        s1 = WMMA_BF16(a_hi, kf3, s1);

        // ----- online softmax (rows = queries; row r lives in one half-wave)
        float mb[8];
#pragma unroll
        for (int v = 0; v < 8; ++v) {
            float x0 = s0[v] * 0.125f;     // 1/sqrt(64)
            float x1 = s1[v] * 0.125f;
            s0[v] = x0; s1[v] = x1;
            mb[v] = fmaxf(x0, x1);
        }
#pragma unroll
        for (int v = 0; v < 8; ++v)
            for (int off = 1; off < 16; off <<= 1)
                mb[v] = fmaxf(mb[v], __shfl_xor(mb[v], off, 32));

        float rs[8];
#pragma unroll
        for (int v = 0; v < 8; ++v) {
            float mn = fmaxf(m_run[v], mb[v]);
            float al = __expf(m_run[v] - mn);
            m_run[v] = mn;
            float p0 = __expf(s0[v] - mn);
            float p1 = __expf(s1[v] - mn);
            rs[v] = p0 + p1;
            l_run[v] *= al;
#pragma unroll
            for (int nt = 0; nt < 4; ++nt) acc[nt][v] *= al;
            // stage P to LDS (C-layout -> row-major [query][key])
            sP[hf * 8 + v][l16]      = (__bf16)p0;
            sP[hf * 8 + v][16 + l16] = (__bf16)p1;
        }
#pragma unroll
        for (int v = 0; v < 8; ++v) {
            for (int off = 1; off < 16; off <<= 1)
                rs[v] += __shfl_xor(rs[v], off, 32);
            l_run[v] += rs[v];
        }

        __syncthreads();   // single wave: cheap; orders LDS store->load

        // ----- O += P @ V : preload P + 4 V-fragments, then 4 WMMAs
        v16bf pa = make_afrag(&sP[l16][0], hf);
        v16bf vf[4];
#pragma unroll
        for (int nt = 0; nt < 4; ++nt)
            vf[nt] = ld_contig16(VT + (size_t)(h * C_D + nt * 16 + l16) * C_MTOT
                                 + rb + kv0 + hf * 16);
#pragma unroll
        for (int nt = 0; nt < 4; ++nt)
            acc[nt] = WMMA_BF16(pa, vf[nt], acc[nt]);

        __syncthreads();
    }

    // ----- normalize and store (row-major bf16 [Mtot, E])
#pragma unroll
    for (int nt = 0; nt < 4; ++nt)
#pragma unroll
        for (int v = 0; v < 8; ++v)
            O[(size_t)(rb + q0 + hf * 8 + v) * C_E + h * C_D + nt * 16 + l16] =
                (__bf16)(acc[nt][v] / l_run[v]);
}

// ---------------------------------------------------------------------------
// Host driver
// ---------------------------------------------------------------------------
extern "C" void kernel_launch(void* const* d_in, const int* in_sizes, int n_in,
                              void* d_out, int out_size, void* d_ws, size_t ws_size,
                              hipStream_t stream) {
    (void)in_sizes; (void)n_in; (void)out_size; (void)ws_size;

    const float* x   = (const float*)d_in[0];
    const float* wq  = (const float*)d_in[1];
    const float* wdn = (const float*)d_in[2];
    const float* wku = (const float*)d_in[3];
    const float* wvu = (const float*)d_in[4];
    const float* wo  = (const float*)d_in[5];
    float* out = (float*)d_out;

    char* ws = (char*)d_ws;
    size_t off = 0;
    auto alloc_bf16 = [&](size_t elems) -> __bf16* {
        __bf16* p = (__bf16*)(ws + off);
        off = (off + elems * sizeof(__bf16) + 255) & ~(size_t)255;
        return p;
    };

    __bf16* xb  = alloc_bf16((size_t)C_MTOT * C_E);
    __bf16* wqb = alloc_bf16((size_t)C_E * C_E);
    __bf16* wdb = alloc_bf16((size_t)C_E * C_L);
    __bf16* wkb = alloc_bf16((size_t)C_L * C_E);
    __bf16* wvb = alloc_bf16((size_t)C_L * C_E);
    __bf16* wob = alloc_bf16((size_t)C_E * C_E);
    __bf16* q   = alloc_bf16((size_t)C_MTOT * C_E);
    __bf16* lat = alloc_bf16((size_t)C_MTOT * C_L);
    __bf16* k   = alloc_bf16((size_t)C_MTOT * C_E);
    __bf16* vT  = alloc_bf16((size_t)C_E * C_MTOT);   // transposed V
    __bf16* ao  = alloc_bf16((size_t)C_MTOT * C_E);

    // fp32 -> bf16 conversions
    auto cvt = [&](const float* s, __bf16* d, int n) {
        cvt_f32_to_bf16<<<dim3((n + 255) / 256), dim3(256), 0, stream>>>(s, d, n);
    };
    cvt(x,   xb,  C_MTOT * C_E);
    cvt(wq,  wqb, C_E * C_E);
    cvt(wdn, wdb, C_E * C_L);
    cvt(wku, wkb, C_L * C_E);
    cvt(wvu, wvb, C_L * C_E);
    cvt(wo,  wob, C_E * C_E);

    const dim3 gthreads(128);
    // q = x @ wq              [4096,1024] = [4096,1024] x [1024,1024]
    gemm_bf16_wmma<<<dim3(C_E / 64, C_MTOT / 128), gthreads, 0, stream>>>(
        xb, wqb, q, C_MTOT, C_E, C_E, 0);
    // lat = x @ w_down        [4096,512]
    gemm_bf16_wmma<<<dim3(C_L / 64, C_MTOT / 128), gthreads, 0, stream>>>(
        xb, wdb, lat, C_MTOT, C_L, C_E, 0);
    // k = lat @ wk_up         [4096,1024]
    gemm_bf16_wmma<<<dim3(C_E / 64, C_MTOT / 128), gthreads, 0, stream>>>(
        lat, wkb, k, C_MTOT, C_E, C_L, 0);
    // vT = (lat @ wv_up)^T    [1024,4096]  (transposed store for attention)
    gemm_bf16_wmma<<<dim3(C_E / 64, C_MTOT / 128), gthreads, 0, stream>>>(
        lat, wvb, vT, C_MTOT, C_E, C_L, 1);

    // RoPE on q and k
    const int pairs = C_MTOT * (C_E / 2);
    rope_inplace<<<dim3((pairs + 255) / 256), dim3(256), 0, stream>>>(q);
    rope_inplace<<<dim3((pairs + 255) / 256), dim3(256), 0, stream>>>(k);

    // paged-causal flash attention -> ao (bf16 [4096,1024])
    flash_attn_wmma<<<dim3(C_T / 16, C_H, C_B), dim3(32), 0, stream>>>(q, k, vT, ao);

    // out = ao @ wo  (f32 output)
    gemm_bf16_wmma<<<dim3(C_E / 64, C_MTOT / 128), gthreads, 0, stream>>>(
        ao, wob, out, C_MTOT, C_E, C_E, 2);
}
